// BitLinearBaseline_60017872994884
// MI455X (gfx1250) — matrix operand order repaired and checked by reading, not compiled, not run on hardware
//
#include <hip/hip_runtime.h>

typedef __bf16 bf16;
typedef __bf16 bf16x8  __attribute__((ext_vector_type(8)));
typedef __bf16 v16bf   __attribute__((ext_vector_type(16)));
typedef float  v8f     __attribute__((ext_vector_type(8)));

#define D_IN   4096
#define D_OUT  4096
#define M_ROWS 8192              // B*S = 4*2048
#define W_ELEMS (4096u*4096u)
#define LN_EPS 1e-5f

// ---------------- workspace layout (bytes) ----------------
// [0      , 4096 )  : 1024 f32 partial sums of w        (mu pass 1)
// [16384  , 32768)  : 4096 f32 partial sums of |w-mu|   (beta pass 1)
// [49152  , 49156)  : mu  scalar
// [49156  , 49160)  : beta scalar
// [65536  , +32MB)  : wbT bf16 [K=4096][N=4096]  (transposed ternary weights)
// [65536+32MB, +64MB): xn bf16 [M=8192][K=4096]  (layernormed activations)

// ---------- kernel 1: per-block partial sums of weight ----------
__global__ void bl_wsum_partial(const float* __restrict__ w, float* __restrict__ part) {
    __shared__ float red[256];
    int t = threadIdx.x;
    size_t base = (size_t)blockIdx.x * 16384;   // 1024 blocks * 16384 = 16.7M
    float s = 0.f;
    #pragma unroll 4
    for (int i = 0; i < 64; ++i) s += w[base + (size_t)i * 256 + t];
    red[t] = s; __syncthreads();
    for (int off = 128; off > 0; off >>= 1) {
        if (t < off) red[t] += red[t + off];
        __syncthreads();
    }
    if (t == 0) part[blockIdx.x] = red[0];
}

// ---------- kernel 2: reduce partials -> scalar (sum * scale) ----------
__global__ void bl_finalize_scalar(const float* __restrict__ part, int n,
                                   float* __restrict__ out, float scale) {
    __shared__ float red[256];
    int t = threadIdx.x;
    float s = 0.f;
    for (int i = t; i < n; i += 256) s += part[i];
    red[t] = s; __syncthreads();
    for (int off = 128; off > 0; off >>= 1) {
        if (t < off) red[t] += red[t + off];
        __syncthreads();
    }
    if (t == 0) out[0] = red[0] * scale;
}

// ---------- kernel 3: binarize + transpose + |w-mu| partials ----------
// grid (64,64): blockIdx.x -> i-tile (K dim), blockIdx.y -> o-tile (N dim)
// block (64,4)
__global__ void bl_binarize(const float* __restrict__ w, const float* __restrict__ mup,
                            bf16* __restrict__ wbT, float* __restrict__ absPart) {
    __shared__ bf16 tile[64][65];
    __shared__ float red[256];
    const float mu = *mup;
    int ix = threadIdx.x;           // position along i within tile
    int oy = threadIdx.y;
    int i0 = blockIdx.x * 64;
    int o0 = blockIdx.y * 64;
    float asum = 0.f;
    #pragma unroll
    for (int r = 0; r < 64; r += 4) {
        int o = o0 + r + oy;
        float d = w[(size_t)o * D_IN + i0 + ix] - mu;
        asum += fabsf(d);
        float s = (d > 0.f) ? 1.f : ((d < 0.f) ? -1.f : 0.f);
        tile[r + oy][ix] = (bf16)s;
    }
    __syncthreads();
    // write transposed: wbT[k][n] = sign(w[n][k]-mu)
    #pragma unroll
    for (int r = 0; r < 64; r += 4) {
        int ii = r + oy;
        wbT[(size_t)(i0 + ii) * D_OUT + o0 + ix] = tile[ix][ii];
    }
    int t = oy * 64 + ix;
    red[t] = asum; __syncthreads();
    for (int off = 128; off > 0; off >>= 1) {
        if (t < off) red[t] += red[t + off];
        __syncthreads();
    }
    if (t == 0) absPart[blockIdx.y * 64 + blockIdx.x] = red[0];
}

// ---------- kernel 4: layernorm row -> bf16 ----------
// one block (256 threads) per row of x [8192 x 4096]
__global__ void bl_layernorm(const float* __restrict__ x, bf16* __restrict__ xn) {
    __shared__ float redS[256];
    __shared__ float redQ[256];
    int t = threadIdx.x;
    size_t row = blockIdx.x;
    const float* xr = x + row * D_IN;
    float v[16];
    float s = 0.f, q = 0.f;
    #pragma unroll
    for (int i = 0; i < 16; ++i) {
        v[i] = xr[i * 256 + t];
        s += v[i];
        q += v[i] * v[i];
    }
    redS[t] = s; redQ[t] = q; __syncthreads();
    for (int off = 128; off > 0; off >>= 1) {
        if (t < off) { redS[t] += redS[t + off]; redQ[t] += redQ[t + off]; }
        __syncthreads();
    }
    float mean = redS[0] * (1.f / D_IN);
    float var  = redQ[0] * (1.f / D_IN) - mean * mean;
    float rstd = rsqrtf(var + LN_EPS);
    bf16* xo = xn + row * D_IN;
    #pragma unroll
    for (int i = 0; i < 16; ++i)
        xo[i * 256 + t] = (bf16)((v[i] - mean) * rstd);
}

// ---------- kernel 5: bf16 WMMA GEMM  out[m,n] = beta * sum_k xn[m,k]*wbT[k,n] ----------
// grid (N/128, M/256); block 256 threads = 8 waves.
// Block tile: 256(M) x 128(N). Wave w owns rows [w*32, w*32+32), all 128 N.
// B K-slabs (64x128) are staged in double-buffered LDS, shared by all 8 waves.
__global__ void __launch_bounds__(256)
bl_gemm_wmma(const bf16* __restrict__ xn, const bf16* __restrict__ wbT,
             const float* __restrict__ betap, float* __restrict__ out) {
    // rows padded to 136 bf16 (68 dwords): per-row bank offset 4 -> optimal
    // 2-way bank occupancy for ds b128 reads; rows stay 16B aligned.
    __shared__ bf16 Bs[2][64][136];

    const int tid  = threadIdx.x;
    const int lane = tid & 31;
    const int wave = tid >> 5;
    const int m0 = blockIdx.y * 256 + wave * 32;
    const int n0 = blockIdx.x * 128;
    const int hi8 = (lane >> 4) * 8;          // A-fragment chunk select per ISA layout

    // staging map: thread -> (row = tid/4 in [0,64), col = (tid%4)*32 elems = 64B)
    const int sRow = tid >> 2;
    const int sCol = (tid & 3) * 32;

    v8f acc[2][8];
    const v8f vzero = {0.f,0.f,0.f,0.f,0.f,0.f,0.f,0.f};
    #pragma unroll
    for (int mi = 0; mi < 2; ++mi)
        #pragma unroll
        for (int ni = 0; ni < 8; ++ni) acc[mi][ni] = vzero;

    const bf16* aRow0 = xn + (size_t)(m0 +      (lane & 15)) * D_IN;
    const bf16* aRow1 = xn + (size_t)(m0 + 16 + (lane & 15)) * D_IN;

    // ---- prologue: stage K-chunk 0 of B into Bs[0] ----
    {
        const bf16* g = wbT + (size_t)sRow * D_OUT + n0 + sCol;
        bf16* s = &Bs[0][sRow][sCol];
        #pragma unroll
        for (int i = 0; i < 4; ++i)
            *(bf16x8*)(s + i * 8) = *(const bf16x8*)(g + i * 8);
    }
    __syncthreads();

    for (int kc = 0; kc < D_IN; kc += 64) {
        const int buf = (kc >> 6) & 1;
        const bool hasNext = (kc + 64) < D_IN;

        // issue global loads for the next B slab early (register double-buffer)
        bf16x8 nx[4];
        if (hasNext) {
            const bf16* g = wbT + (size_t)(kc + 64 + sRow) * D_OUT + n0 + sCol;
            #pragma unroll
            for (int i = 0; i < 4; ++i) nx[i] = *(const bf16x8*)(g + i * 8);
            // global_prefetch_b8, WGP scope (pull into all cache levels): A stream
            __builtin_prefetch(aRow0 + kc + 64, 0, 3);
        }

        // ---- compute on Bs[buf]: two 32-K steps, 16 WMMA each ----
        #pragma unroll
        for (int ks = 0; ks < 64; ks += 32) {
            const int k0 = kc + ks;
            // A fragments: 16x32 bf16; lanes 0-15 hold K 0..7 & 16..23,
            // lanes 16-31 hold K 8..15 & 24..31 (per ISA A layout)
            v16bf a[2];
            {
                bf16x8 lo = *(const bf16x8*)(aRow0 + k0 + hi8);
                bf16x8 hh = *(const bf16x8*)(aRow0 + k0 + 16 + hi8);
                a[0] = __builtin_shufflevector(lo, hh, 0,1,2,3,4,5,6,7,8,9,10,11,12,13,14,15);
                lo = *(const bf16x8*)(aRow1 + k0 + hi8);
                hh = *(const bf16x8*)(aRow1 + k0 + 16 + hi8);
                a[1] = __builtin_shufflevector(lo, hh, 0,1,2,3,4,5,6,7,8,9,10,11,12,13,14,15);
            }
            // B fragments: 32x16; lane L holds N=L%16, K same as A layout
            // (lanes 0-15: K 0..7 & 16..23, lanes 16-31: K 8..15 & 24..31) (LDS).
            // Software-pipeline the LDS reads one fragment ahead so ds_load
            // latency overlaps the in-flight WMMA pair.
            const bf16* bsBase = &Bs[buf][ks + hi8][lane & 15];
            v16bf bcur;
            {
                #pragma unroll
                for (int s = 0; s < 16; ++s)
                    bcur[s] = bsBase[((s & 7) + ((s >> 3) << 4)) * 136];
            }
            #pragma unroll
            for (int ni = 0; ni < 8; ++ni) {
                v16bf bnext = bcur;
                if (ni < 7) {
                    const bf16* bs = bsBase + (ni + 1) * 16;
                    #pragma unroll
                    for (int s = 0; s < 16; ++s)
                        bnext[s] = bs[((s & 7) + ((s >> 3) << 4)) * 136];
                }
                acc[0][ni] = __builtin_amdgcn_wmma_f32_16x16x32_bf16(
                    false, a[0], false, bcur, (short)0, acc[0][ni], false, false);
                acc[1][ni] = __builtin_amdgcn_wmma_f32_16x16x32_bf16(
                    false, a[1], false, bcur, (short)0, acc[1][ni], false, false);
                bcur = bnext;
            }
        }

        // store the prefetched slab into the other LDS buffer
        if (hasNext) {
            bf16* s = &Bs[buf ^ 1][sRow][sCol];
            #pragma unroll
            for (int i = 0; i < 4; ++i) *(bf16x8*)(s + i * 8) = nx[i];
        }
        __syncthreads();
    }

    const float beta = *betap;
    // C/D layout: VGPR j -> lanes 0-15: M=j, lanes 16-31: M=8+j; N = lane%16
    const int colBase = n0 + (lane & 15);
    const int rowBase = m0 + ((lane >> 4) << 3);
    #pragma unroll
    for (int mi = 0; mi < 2; ++mi)
        #pragma unroll
        for (int ni = 0; ni < 8; ++ni)
            #pragma unroll
            for (int j = 0; j < 8; ++j)
                out[(size_t)(rowBase + mi * 16 + j) * D_OUT + colBase + ni * 16] =
                    acc[mi][ni][j] * beta;
}

extern "C" void kernel_launch(void* const* d_in, const int* in_sizes, int n_in,
                              void* d_out, int out_size, void* d_ws, size_t ws_size,
                              hipStream_t stream) {
    const float* x = (const float*)d_in[0];      // [4,2048,4096] f32
    const float* w = (const float*)d_in[1];      // [4096,4096]  f32
    float* out = (float*)d_out;                  // [4,2048,4096] f32

    char* ws = (char*)d_ws;
    float* sumPart = (float*)(ws + 0);           // 1024 f32
    float* absPart = (float*)(ws + 16384);       // 4096 f32
    float* mu      = (float*)(ws + 49152);
    float* beta    = (float*)(ws + 49156);
    bf16*  wbT     = (bf16*)(ws + 65536);                        // 32 MB
    bf16*  xn      = (bf16*)(ws + 65536 + (size_t)32*1024*1024); // 64 MB

    const float invW = 1.0f / (float)W_ELEMS;

    // 1) mu = mean(w)
    bl_wsum_partial<<<1024, 256, 0, stream>>>(w, sumPart);
    bl_finalize_scalar<<<1, 256, 0, stream>>>(sumPart, 1024, mu, invW);
    // 2) wbT = sign(w - mu)^T (bf16), partials of |w - mu|
    bl_binarize<<<dim3(64, 64), dim3(64, 4), 0, stream>>>(w, mu, wbT, absPart);
    bl_finalize_scalar<<<1, 256, 0, stream>>>(absPart, 4096, beta, invW);
    // 3) xn = layernorm(x) (bf16)
    bl_layernorm<<<M_ROWS, 256, 0, stream>>>(x, xn);
    // 4) out = beta * xn @ wbT  (M=8192, N=4096, K=4096)
    bl_gemm_wmma<<<dim3(D_OUT / 128, M_ROWS / 256), 256, 0, stream>>>(xn, wbT, beta, out);
}
